// self_attention_26628797236080
// MI455X (gfx1250) — matrix operand order repaired and checked by reading, not compiled, not run on hardware
//
#include <hip/hip_runtime.h>
#include <hip/hip_bf16.h>

// ---------------------------------------------------------------------------
// Multi-head self-attention for MI455X (gfx1250, wave32, WMMA f32_16x16x32_f16)
//   B=4, S=2048, D_MODEL=1024, HEADS=16, D_K=64
// Pipeline:
//   1) proj_kernel   : Q/K/V per-head projection (WMMA GEMM, f32 in -> f16 out)
//                      V written feature-major [b,h,k,s] for the P*V B-operand
//   2) cvt_kernel    : Wo f32 -> f16
//   3) attn_kernel   : flash-attention; 4 waves/block share one (b,h) and
//                      cooperatively double-buffer K/V tiles in LDS (4x L2
//                      traffic reduction), online softmax via shfl_xor,
//                      P relayout through LDS, all matmuls via WMMA.
//   4) outproj_kernel: [B*S,1024] @ Wo^T + bo (WMMA GEMM, f32 out)
// ---------------------------------------------------------------------------

#define BATCH   4
#define SEQ     2048
#define NHEADS  16
#define DK      64
#define DM      1024

typedef __attribute__((ext_vector_type(16))) _Float16 v16h;
typedef __attribute__((ext_vector_type(8)))  _Float16 v8h;
typedef __attribute__((ext_vector_type(8)))  float    v8f;

static __device__ __forceinline__ v16h cat8(v8h lo, v8h hi) {
  return __builtin_shufflevector(lo, hi, 0,1,2,3,4,5,6,7,8,9,10,11,12,13,14,15);
}

static __device__ __forceinline__ v8f wmma16(v16h a, v16h b, v8f c) {
  // D = A(16x32 f16) * B(32x16 f16) + C(16x16 f32)
  return __builtin_amdgcn_wmma_f32_16x16x32_f16(false, a, false, b,
                                                (short)0, c, false, false);
}

// ---------------------------------------------------------------------------
// 1) Per-head projection: rows are (b,s,h) triples; 16 consecutive rows share
//    one token (h = row % 16). One wave computes a 16x64 output tile:
//    out[m][k] = sum_j X[m][j] * W[k][j]   (x @ W^T)
// vTranspose=0 -> Out[b,h,s,k] ; vTranspose=1 -> Out[b,h,k,s]
// ---------------------------------------------------------------------------
__global__ void __launch_bounds__(128)
proj_kernel(const float* __restrict__ X, const float* __restrict__ W,
            _Float16* __restrict__ Out, int vTranspose) {
  __shared__ __attribute__((aligned(32))) _Float16 Wl[DK * DK]; // W[n][k], f16
  const int tid = threadIdx.x;
  for (int i = tid; i < DK * DK; i += 128) Wl[i] = (_Float16)W[i];
  __syncthreads();

  const int lane = tid & 31;
  const int warp = tid >> 5;
  const int rowTile = blockIdx.x * 4 + warp;      // == (b*SEQ + s)
  const int r      = lane & 15;
  const int kbaseA = (lane < 16) ? 0 : 16;        // A operand K base (contiguous, matches B)
  const int kkoff  = (lane < 16) ? 0 : 16;        // B operand K base
  const int hiRow  = (lane < 16) ? 0 : 8;         // C layout row offset

  // A operand: row (head) = lane&15, 64 f32 contiguous -> two 16x32 f16 chunks
  const float* xrow = X + ((size_t)rowTile * 16 + r) * DK;
  v16h a[2];
  #pragma unroll
  for (int c = 0; c < 2; ++c) {
    const int K0 = c * 32;
    #pragma unroll
    for (int i = 0; i < 8; ++i) {
      a[c][i]     = (_Float16)xrow[K0 + kbaseA + i];
      a[c][8 + i] = (_Float16)xrow[K0 + kbaseA + 8 + i];
    }
  }

  v8f acc[4] = {};                                 // 4 N-tiles of 16
  #pragma unroll
  for (int nt = 0; nt < 4; ++nt) {
    #pragma unroll
    for (int c = 0; c < 2; ++c) {
      // B[kk][n] = W[n][kk] : contiguous 16 halves of row n
      v16h b = *(const v16h*)&Wl[(size_t)(nt * 16 + (lane & 15)) * DK + c * 32 + kkoff];
      acc[nt] = wmma16(a[c], b, acc[nt]);
    }
  }

  const int b  = rowTile / SEQ;
  const int s  = rowTile % SEQ;
  #pragma unroll
  for (int nt = 0; nt < 4; ++nt) {
    #pragma unroll
    for (int i = 0; i < 8; ++i) {
      const int h = i + hiRow;                    // C row == head index
      const int k = nt * 16 + (lane & 15);        // C col == feature
      const size_t dst = vTranspose
          ? (((size_t)(b * NHEADS + h) * DK + k) * SEQ + s)
          : (((size_t)(b * NHEADS + h) * SEQ + s) * DK + k);
      Out[dst] = (_Float16)acc[nt][i];
    }
  }
}

// ---------------------------------------------------------------------------
// 2) Wo f32 -> f16
// ---------------------------------------------------------------------------
__global__ void cvt_kernel(const float* __restrict__ W, _Float16* __restrict__ O, int n) {
  int i = blockIdx.x * 256 + threadIdx.x;
  if (i < n) O[i] = (_Float16)W[i];
}

// ---------------------------------------------------------------------------
// 3) Flash attention. Block = 4 waves, all on the same (b,h); wave w owns
//    queries [q0 + 16w, q0 + 16w + 16). Each 32-key iteration the block
//    cooperatively stages K (32x64, row-major) and V (64x32, feature-major)
//    into a double-buffered LDS tile: global K/V is read ONCE per 64 queries
//    instead of once per 16 (4x less L2 traffic). One barrier per iteration.
// ---------------------------------------------------------------------------
__global__ void __launch_bounds__(128)
attn_kernel(const _Float16* __restrict__ Qh, const _Float16* __restrict__ Kh,
            const _Float16* __restrict__ Vt, const int* __restrict__ mask,
            _Float16* __restrict__ AttnOut) {
  __shared__ __attribute__((aligned(32))) _Float16 Kl[2][32 * DK];  // [key][feat]
  __shared__ __attribute__((aligned(32))) _Float16 Vl[2][DK * 32];  // [feat][key]
  __shared__ __attribute__((aligned(32))) _Float16 Pbuf[4][16 * 32];

  const int tid  = threadIdx.x;
  const int lane = tid & 31;
  const int warp = tid >> 5;
  const int qBlocks = SEQ / 64;                    // 64-query groups per (b,h)
  const int bh  = blockIdx.x / qBlocks;            // b*NHEADS + h
  const int q0  = (blockIdx.x % qBlocks) * 64 + warp * 16;

  const _Float16* Qb = Qh + (size_t)bh * SEQ * DK;
  const _Float16* Kb = Kh + (size_t)bh * SEQ * DK;
  const _Float16* Vb = Vt + (size_t)bh * DK * SEQ; // [feature][s]

  const int r      = lane & 15;
  const int kbaseA = (lane < 16) ? 0 : 16;
  const int kkoff  = (lane < 16) ? 0 : 16;
  const int hiRow  = (lane < 16) ? 0 : 8;

  // Cooperative staging indices (128 threads move 4KB K + 4KB V per tile)
  const int kKey = tid >> 2;                       // 0..31
  const int kFo  = (tid & 3) * 16;                 // feature offset 0/16/32/48
  const int vF   = tid >> 1;                       // 0..63
  const int vKo  = (tid & 1) * 16;                 // key offset 0/16

  // Q A-operand: 16x64 as two 16x32 chunks (v8h pairs, contiguous)
  const _Float16* qrow = Qb + (size_t)(q0 + r) * DK;
  v16h qa[2];
  #pragma unroll
  for (int c = 0; c < 2; ++c) {
    v8h lo = *(const v8h*)&qrow[c * 32 + kbaseA];
    v8h hi = *(const v8h*)&qrow[c * 32 + kbaseA + 8];
    qa[c] = cat8(lo, hi);
  }

  v8f o[4] = {};                                   // O accumulator, 16x64
  float mcur[8], lsum[8];
  #pragma unroll
  for (int i = 0; i < 8; ++i) { mcur[i] = -3.0e38f; lsum[i] = 0.0f; }

  _Float16* pb = Pbuf[warp];
  const float scl = 0.125f;                        // 1/sqrt(64)
  const int NITER = SEQ / 32;

  // Prologue: stage tile 0 into buffer 0
  *(v16h*)&Kl[0][kKey * DK + kFo] = *(const v16h*)&Kb[(size_t)kKey * DK + kFo];
  *(v16h*)&Vl[0][vF * 32 + vKo]   = *(const v16h*)&Vb[(size_t)vF * SEQ + vKo];

  for (int j = 0; j < NITER; ++j) {
    const int buf = j & 1;
    const int kt0 = j * 32;
    __syncthreads();                               // tile j visible; buf^1 free

    if (j + 1 < NITER) {
      const int nt0 = kt0 + 32;
      *(v16h*)&Kl[buf ^ 1][kKey * DK + kFo] =
          *(const v16h*)&Kb[(size_t)(nt0 + kKey) * DK + kFo];
      *(v16h*)&Vl[buf ^ 1][vF * 32 + vKo] =
          *(const v16h*)&Vb[(size_t)vF * SEQ + nt0 + vKo];
    }
    if (j + 2 < NITER) {
      // global_prefetch_b8: pull tile j+2 of the K/V streams toward L2/L0
      __builtin_prefetch(&Kb[(size_t)(kt0 + 64 + kKey) * DK + kFo], 0, 3);
      __builtin_prefetch(&Vb[(size_t)vF * SEQ + kt0 + 64 + vKo], 0, 3);
    }

    // ---- S = Q K^T for 32 keys (two 16x16 C tiles), operands from LDS ----
    v8f s0 = {}, s1 = {};
    #pragma unroll
    for (int c = 0; c < 2; ++c) {
      v16h kb0 = *(const v16h*)&Kl[buf][(lane & 15) * DK + c * 32 + kkoff];
      v16h kb1 = *(const v16h*)&Kl[buf][((lane & 15) + 16) * DK + c * 32 + kkoff];
      s0 = wmma16(qa[c], kb0, s0);
      s1 = wmma16(qa[c], kb1, s1);
    }

    // ---- scale + mask ----
    #pragma unroll
    for (int i = 0; i < 8; ++i) {
      const int qg = q0 + i + hiRow;
      const size_t mrow = (size_t)qg * SEQ + kt0 + (lane & 15);
      float v0 = s0[i] * scl;
      float v1 = s1[i] * scl;
      if (mask[mrow] == 0)      v0 = -1.0e30f;
      if (mask[mrow + 16] == 0) v1 = -1.0e30f;
      s0[i] = v0; s1[i] = v1;
    }

    // ---- online softmax (per C-layout row; rows live in 16-lane halves) ----
    float rscale[8];
    #pragma unroll
    for (int i = 0; i < 8; ++i) {
      float t = fmaxf(s0[i], s1[i]);
      #pragma unroll
      for (int off = 1; off < 16; off <<= 1) t = fmaxf(t, __shfl_xor(t, off, 16));
      const float mnew = fmaxf(mcur[i], t);
      const float sc   = __expf(mcur[i] - mnew);
      const float p0   = __expf(s0[i] - mnew);
      const float p1   = __expf(s1[i] - mnew);
      float rs = p0 + p1;
      #pragma unroll
      for (int off = 1; off < 16; off <<= 1) rs += __shfl_xor(rs, off, 16);
      lsum[i] = lsum[i] * sc + rs;
      mcur[i] = mnew;
      rscale[i] = sc;
      s0[i] = p0; s1[i] = p1;
    }
    #pragma unroll
    for (int nt = 0; nt < 4; ++nt)
      #pragma unroll
      for (int i = 0; i < 8; ++i) o[nt][i] *= rscale[i];

    // ---- C-layout P -> LDS -> A-layout (16x32 f16); wave-private region ----
    #pragma unroll
    for (int i = 0; i < 8; ++i) {
      const int row = i + hiRow, col = lane & 15;
      pb[row * 32 + col]      = (_Float16)s0[i];
      pb[row * 32 + col + 16] = (_Float16)s1[i];
    }
    v8h plo = *(const v8h*)&pb[r * 32 + kbaseA];
    v8h phi = *(const v8h*)&pb[r * 32 + kbaseA + 8];
    v16h pa = cat8(plo, phi);

    // ---- O += P * V  (B operand contiguous: feature-major V tile) ----
    #pragma unroll
    for (int nt = 0; nt < 4; ++nt) {
      v16h vb = *(const v16h*)&Vl[buf][(nt * 16 + (lane & 15)) * 32 + kkoff];
      o[nt] = wmma16(pa, vb, o[nt]);
    }
  }

  // ---- normalize and store as f16 [b, s, h*64 + k] for the output GEMM ----
  const int b = bh / NHEADS;
  const int h = bh % NHEADS;
  #pragma unroll
  for (int nt = 0; nt < 4; ++nt) {
    #pragma unroll
    for (int i = 0; i < 8; ++i) {
      const int qg   = q0 + i + hiRow;
      const int feat = nt * 16 + (lane & 15);
      const float val = o[nt][i] / lsum[i];
      AttnOut[((size_t)(b * SEQ + qg)) * DM + h * DK + feat] = (_Float16)val;
    }
  }
}

// ---------------------------------------------------------------------------
// 4) Output projection: Out[m][n] = sum_k A[m][k]*Wo[n][k] + bo[n]
//    M = B*S = 8192, N = K = 1024. One wave -> 16x64 tile, K-loop of 32.
//    Wo16 is 2 MB f16 with 512-way reuse -> L2 resident.
// ---------------------------------------------------------------------------
__global__ void __launch_bounds__(128)
outproj_kernel(const _Float16* __restrict__ A, const _Float16* __restrict__ Wo16,
               const float* __restrict__ bo, float* __restrict__ Out) {
  const int tid  = threadIdx.x;
  const int lane = tid & 31;
  const int warp = tid >> 5;
  const int t    = blockIdx.x * 4 + warp;
  const int nt64 = t % (DM / 64);
  const int mt   = t / (DM / 64);
  const int m0 = mt * 16, n0 = nt64 * 64;

  const int r      = lane & 15;
  const int kbaseA = (lane < 16) ? 0 : 16;
  const int kkoff  = (lane < 16) ? 0 : 16;
  const int hiRow  = (lane < 16) ? 0 : 8;

  v8f acc[4] = {};
  for (int K0 = 0; K0 < DM; K0 += 32) {
    const _Float16* arow = A + (size_t)(m0 + r) * DM + K0;
    v8h alo = *(const v8h*)&arow[kbaseA];
    v8h ahi = *(const v8h*)&arow[kbaseA + 8];
    v16h av = cat8(alo, ahi);
    #pragma unroll
    for (int nt = 0; nt < 4; ++nt) {
      v16h bv = *(const v16h*)&Wo16[(size_t)(n0 + nt * 16 + (lane & 15)) * DM + K0 + kkoff];
      acc[nt] = wmma16(av, bv, acc[nt]);
    }
  }

  #pragma unroll
  for (int nt = 0; nt < 4; ++nt) {
    const float bias = bo[n0 + nt * 16 + (lane & 15)];
    #pragma unroll
    for (int i = 0; i < 8; ++i) {
      const int m = m0 + i + hiRow;
      Out[(size_t)m * DM + n0 + nt * 16 + (lane & 15)] = acc[nt][i] + bias;
    }
  }
}

// ---------------------------------------------------------------------------
// Launch
// ---------------------------------------------------------------------------
extern "C" void kernel_launch(void* const* d_in, const int* in_sizes, int n_in,
                              void* d_out, int out_size, void* d_ws, size_t ws_size,
                              hipStream_t stream) {
  (void)in_sizes; (void)n_in; (void)out_size; (void)ws_size;
  const float* values = (const float*)d_in[0];
  const float* keys_  = (const float*)d_in[1];
  const float* query  = (const float*)d_in[2];
  const int*   mask   = (const int*)d_in[3];
  const float* Wv     = (const float*)d_in[4];
  const float* Wk     = (const float*)d_in[5];
  const float* Wq     = (const float*)d_in[6];
  const float* Wo     = (const float*)d_in[7];
  const float* bo     = (const float*)d_in[8];
  float* out = (float*)d_out;

  // Workspace carve-up (f16): Qh, Kh (both [b,h,s,k]), Vt ([b,h,k,s]),
  // AttnOut ([b,s,h*k]), Wo16. Total ~66 MB.
  const size_t qkvElems = (size_t)BATCH * NHEADS * SEQ * DK; // 8,388,608
  _Float16* Qh   = (_Float16*)d_ws;
  _Float16* Kh   = Qh + qkvElems;
  _Float16* Vt   = Kh + qkvElems;
  _Float16* At   = Vt + qkvElems;
  _Float16* Wo16 = At + qkvElems;

  dim3 blk(128);
  const int projBlocks = (BATCH * SEQ) / 4;                     // 2048 (4 waves/blk)
  proj_kernel<<<projBlocks, blk, 0, stream>>>(query,  Wq, Qh, 0);
  proj_kernel<<<projBlocks, blk, 0, stream>>>(keys_,  Wk, Kh, 0);
  proj_kernel<<<projBlocks, blk, 0, stream>>>(values, Wv, Vt, 1);
  cvt_kernel<<<(DM * DM + 255) / 256, 256, 0, stream>>>(Wo, Wo16, DM * DM);

  const int attnBlocks = BATCH * NHEADS * (SEQ / 64);           // 2048
  attn_kernel<<<attnBlocks, blk, 0, stream>>>(Qh, Kh, Vt, mask, At);

  const int gemmBlocks = ((BATCH * SEQ / 16) * (DM / 64)) / 4;  // 2048
  outproj_kernel<<<gemmBlocks, blk, 0, stream>>>(At, Wo16, bo, out);
}
